// RNNModel_52587579572275
// MI455X (gfx1250) — matrix operand order repaired and checked by reading, not compiled, not run on hardware
//
#include <hip/hip_runtime.h>
#include <hip/hip_bf16.h>

#define T_STEPS 512
#define BATCH   64
#define HID     1024
#define NWG     32          // workgroups in persistent scan kernel
#define COLS_WG 32          // output columns per workgroup in scan

typedef __attribute__((ext_vector_type(16))) __bf16 v16bf;
typedef __attribute__((ext_vector_type(8)))  float  v8f;

union Frag {
    uint4 u[2];
    v16bf v;
};

__device__ __forceinline__ unsigned short f2bf(float f) {
    unsigned u = __float_as_uint(f);
    unsigned r = u + 0x7FFFu + ((u >> 16) & 1u);   // round-to-nearest-even
    return (unsigned short)(r >> 16);
}

// ---------------------------------------------------------------- conversion
__global__ __launch_bounds__(256) void cvt_f32_bf16(const float* __restrict__ in,
                                                    unsigned short* __restrict__ out,
                                                    int n) {
    int i = blockIdx.x * blockDim.x + threadIdx.x;
    if (i < n) out[i] = f2bf(in[i]);
}

// ---------------------------------------------------------------- xp GEMM
// xp[t*B + b][n] = sum_k X[b][t][k] * W[n][k] + b_ih[n] + b_hh[n]
// X bf16 [B][T][H]; W bf16 [H][H] (torch layout W[out][in]); xp fp32 [T*B][H].
// WG tile: 64M x 128N, K-slab 128 staged in LDS; 8 waves, each 64M x 16N.
// All 4 A fragments are loaded into distinct registers before the 4 WMMAs so
// the matrix pipe overlaps with ds_load instead of waiting dscnt==0 per op.
__global__ __launch_bounds__(256) void gemm_xp_kernel(
        const unsigned short* __restrict__ Xbf,
        const unsigned short* __restrict__ Wbf,
        const float* __restrict__ b_ih,
        const float* __restrict__ b_hh,
        float* __restrict__ xp) {
    __shared__ unsigned short Asl[64][128];     // 16 KB A slab (bf16)

    const int lane = threadIdx.x & 31;
    const int wave = threadIdx.x >> 5;          // 0..7 -> N tile
    const int l16  = lane & 15;
    const int hi   = lane >> 4;                 // 0 or 1
    const int m0   = blockIdx.y * 64;           // row base in [T*B)
    const int n    = blockIdx.x * 128 + wave * 16 + l16;

    const unsigned short* brow = Wbf + (size_t)n * HID;
    const int kbA = hi ? 8 : 0;     // A: lanes16-31 hold K[8..16)+[24..32)
    const int kbB = hi ? 16 : 0;    // unused: B is loaded with kbA, same k order as A

    v8f acc[4] = {};

    for (int ks = 0; ks < HID; ks += 128) {
        __syncthreads();
        // Cooperative A slab load: 64 rows x 128 halfs = 1024 uint4,
        // 4 per thread. Row tb -> x address via (b,t) de-interleave.
#pragma unroll
        for (int j = 0; j < 4; ++j) {
            int idx = threadIdx.x + j * 256;    // uint4 index in slab
            int r   = idx >> 4;                 // row 0..63
            int c   = (idx & 15) << 3;          // half offset 0..120
            int tb  = m0 + r;
            const unsigned short* src =
                Xbf + (size_t)((tb & 63) * T_STEPS + (tb >> 6)) * HID + ks + c;
            *(uint4*)&Asl[r][c] = *(const uint4*)src;
        }
        __syncthreads();

#pragma unroll
        for (int kc = 0; kc < 128; kc += 32) {
            Frag w;
            w.u[0] = *(const uint4*)(brow + ks + kc + kbA);
            w.u[1] = *(const uint4*)(brow + ks + kc + kbA + 16);

            Frag a0, a1, a2, a3;                 // distinct register sets
            const unsigned short* ar0 = &Asl[ 0 + l16][kc];
            const unsigned short* ar1 = &Asl[16 + l16][kc];
            const unsigned short* ar2 = &Asl[32 + l16][kc];
            const unsigned short* ar3 = &Asl[48 + l16][kc];
            a0.u[0] = *(const uint4*)(ar0 + kbA);          // 8x ds_load_b128
            a0.u[1] = *(const uint4*)(ar0 + kbA + 16);
            a1.u[0] = *(const uint4*)(ar1 + kbA);
            a1.u[1] = *(const uint4*)(ar1 + kbA + 16);
            a2.u[0] = *(const uint4*)(ar2 + kbA);
            a2.u[1] = *(const uint4*)(ar2 + kbA + 16);
            a3.u[0] = *(const uint4*)(ar3 + kbA);
            a3.u[1] = *(const uint4*)(ar3 + kbA + 16);

            acc[0] = __builtin_amdgcn_wmma_f32_16x16x32_bf16(
                    false, a0.v, false, w.v, (short)0, acc[0], false, false);
            acc[1] = __builtin_amdgcn_wmma_f32_16x16x32_bf16(
                    false, a1.v, false, w.v, (short)0, acc[1], false, false);
            acc[2] = __builtin_amdgcn_wmma_f32_16x16x32_bf16(
                    false, a2.v, false, w.v, (short)0, acc[2], false, false);
            acc[3] = __builtin_amdgcn_wmma_f32_16x16x32_bf16(
                    false, a3.v, false, w.v, (short)0, acc[3], false, false);
        }
    }

    const float bias = b_ih[n] + b_hh[n];
#pragma unroll
    for (int mt = 0; mt < 4; ++mt) {
#pragma unroll
        for (int r = 0; r < 8; ++r) {
            int m = m0 + mt * 16 + r + hi * 8;
            xp[(size_t)m * HID + n] = acc[mt][r] + bias;
        }
    }
}

// ---------------------------------------------------------------- grid barrier
__device__ __forceinline__ void grid_barrier(unsigned* counter, unsigned* gen,
                                             unsigned nwg, unsigned* localGen) {
    __syncthreads();
    if (threadIdx.x == 0) {
        __threadfence();   // make this WG's stores agent-visible
        unsigned g = *localGen;
        unsigned arrive = __hip_atomic_fetch_add(counter, 1u, __ATOMIC_ACQ_REL,
                                                 __HIP_MEMORY_SCOPE_AGENT);
        if (arrive == nwg - 1u) {
            __hip_atomic_store(counter, 0u, __ATOMIC_RELAXED, __HIP_MEMORY_SCOPE_AGENT);
            __hip_atomic_store(gen, g + 1u, __ATOMIC_RELEASE, __HIP_MEMORY_SCOPE_AGENT);
        } else {
            while (__hip_atomic_load(gen, __ATOMIC_ACQUIRE,
                                     __HIP_MEMORY_SCOPE_AGENT) == g) {
                __builtin_amdgcn_s_sleep(2);
            }
        }
        *localGen = g + 1u;
    }
    __syncthreads();
}

// ---------------------------------------------------------------- persistent scan
// h_{t} = tanh(xp[t] + h_{t-1} @ W_hh^T), h_{-1}=0.  xp fp32 [T][B][H].
// Writes per-timestep hidden state to out_bf (bf16, [B][T][H]) or out_f32.
__global__ __launch_bounds__(256) void rnn_scan_kernel(
        const float* __restrict__ xp,
        const float* __restrict__ Whh,          // fp32 [H][H]
        unsigned short* __restrict__ hstate,    // [2][B][H] bf16 ping-pong
        unsigned short* __restrict__ out_bf,    // [B][T][H] or nullptr
        float* __restrict__ out_f32,            // [B][T][H] or nullptr
        unsigned* __restrict__ syncArea) {      // [0]=counter, [16]=gen
    __shared__ unsigned short Wl[COLS_WG][HID]; // 64 KB W_hh column slice (bf16)

    const int wg   = blockIdx.x;
    const int lane = threadIdx.x & 31;
    const int wave = threadIdx.x >> 5;      // 0..7
    const int mB   = wave & 3;              // batch block (4 x 16 = 64)
    const int nBl  = wave >> 2;             // 0..1 local col block
    const int l16  = lane & 15;
    const int hi   = lane >> 4;
    const int kbA  = hi ? 8 : 0;
    const int kbB  = hi ? 16 : 0;

    // Stage this WG's W_hh column slice into LDS (fp32 -> bf16).
    const int col0 = wg * COLS_WG;
    for (int i = threadIdx.x; i < COLS_WG * HID; i += blockDim.x) {
        int c = i >> 10;
        int k = i & (HID - 1);
        Wl[c][k] = f2bf(Whh[(size_t)(col0 + c) * HID + k]);
    }
    // Zero initial hidden state (buffer 0), distributed over the whole grid.
    for (int i = blockIdx.x * blockDim.x + threadIdx.x; i < BATCH * HID;
         i += gridDim.x * blockDim.x) {
        hstate[i] = 0;
    }

    unsigned localGen = 0;
    unsigned* counter = syncArea;
    unsigned* gen     = syncArea + 16;
    grid_barrier(counter, gen, NWG, &localGen);   // h0 visible everywhere

    const int ncol = col0 + nBl * 16 + l16;
    const unsigned short* brow = &Wl[nBl * 16 + l16][0];

    for (int t = 0; t < T_STEPS; ++t) {
        const unsigned short* hcur = hstate + (size_t)(t & 1) * (BATCH * HID);
        unsigned short*       hnxt = hstate + (size_t)((t & 1) ^ 1) * (BATCH * HID);
        const unsigned short* arow = hcur + (size_t)(mB * 16 + l16) * HID;

        v8f acc = {};
#pragma unroll 4
        for (int k0 = 0; k0 < HID; k0 += 32) {
            Frag a, w;
            a.u[0] = *(const uint4*)(arow + k0 + kbA);
            a.u[1] = *(const uint4*)(arow + k0 + kbA + 16);
            w.u[0] = *(const uint4*)(brow + k0 + kbA);     // ds_load_b128
            w.u[1] = *(const uint4*)(brow + k0 + kbA + 16);
            acc = __builtin_amdgcn_wmma_f32_16x16x32_bf16(
                    false, a.v, false, w.v, (short)0, acc, false, false);
        }

        const float* xprow = xp + (size_t)t * (BATCH * HID);
#pragma unroll
        for (int r = 0; r < 8; ++r) {
            int m = mB * 16 + r + hi * 8;                  // batch index
            float v = tanhf(acc[r] + xprow[(size_t)m * HID + ncol]);
            hnxt[(size_t)m * HID + ncol] = f2bf(v);
            size_t oidx = ((size_t)m * T_STEPS + t) * HID + ncol;
            if (out_f32) out_f32[oidx] = v;
            else         out_bf[oidx] = f2bf(v);
        }
        grid_barrier(counter, gen, NWG, &localGen);
    }
}

// ---------------------------------------------------------------- launch
extern "C" void kernel_launch(void* const* d_in, const int* in_sizes, int n_in,
                              void* d_out, int out_size, void* d_ws, size_t ws_size,
                              hipStream_t stream) {
    const float* x     = (const float*)d_in[0];
    const float* W_ih0 = (const float*)d_in[1];
    const float* W_hh0 = (const float*)d_in[2];
    const float* b_ih0 = (const float*)d_in[3];
    const float* b_hh0 = (const float*)d_in[4];
    const float* W_ih1 = (const float*)d_in[5];
    const float* W_hh1 = (const float*)d_in[6];
    const float* b_ih1 = (const float*)d_in[7];
    const float* b_hh1 = (const float*)d_in[8];
    float* out = (float*)d_out;

    char* ws = (char*)d_ws;
    size_t off = 0;
    auto alloc = [&](size_t bytes) {
        size_t o = off;
        off += (bytes + 255) & ~(size_t)255;
        return o;
    };
    unsigned*       syncArea = (unsigned*)(ws + alloc(256));
    unsigned short* x_bf     = (unsigned short*)(ws + alloc((size_t)BATCH * T_STEPS * HID * 2));
    unsigned short* wih0_bf  = (unsigned short*)(ws + alloc((size_t)HID * HID * 2));
    unsigned short* wih1_bf  = (unsigned short*)(ws + alloc((size_t)HID * HID * 2));
    float*          xp       = (float*)(ws + alloc((size_t)T_STEPS * BATCH * HID * 4));
    unsigned short* h1_bf    = (unsigned short*)(ws + alloc((size_t)BATCH * T_STEPS * HID * 2));
    unsigned short* hstate   = (unsigned short*)(ws + alloc((size_t)2 * BATCH * HID * 2));

    hipMemsetAsync(syncArea, 0, 256, stream);

    // fp32 -> bf16 conversions
    {
        int n = BATCH * T_STEPS * HID;
        cvt_f32_bf16<<<(n + 255) / 256, 256, 0, stream>>>(x, x_bf, n);
        n = HID * HID;
        cvt_f32_bf16<<<(n + 255) / 256, 256, 0, stream>>>(W_ih0, wih0_bf, n);
        cvt_f32_bf16<<<(n + 255) / 256, 256, 0, stream>>>(W_ih1, wih1_bf, n);
    }

    dim3 ggrid(HID / 128, (T_STEPS * BATCH) / 64);

    // -------- layer 0
    gemm_xp_kernel<<<ggrid, 256, 0, stream>>>(x_bf, wih0_bf, b_ih0, b_hh0, xp);
    rnn_scan_kernel<<<NWG, 256, 0, stream>>>(xp, W_hh0, hstate, h1_bf, nullptr,
                                             syncArea);
    // -------- layer 1
    gemm_xp_kernel<<<ggrid, 256, 0, stream>>>(h1_bf, wih1_bf, b_ih1, b_hh1, xp);
    rnn_scan_kernel<<<NWG, 256, 0, stream>>>(xp, W_hh1, hstate, nullptr, out,
                                             syncArea + 32);
}
